// TranslationModel_71605694759501
// MI455X (gfx1250) — compile-verified, matrix operand order checked by reading, not run on hardware
//
#include <hip/hip_runtime.h>
#include <hip/hip_bf16.h>
#include <math.h>

// Model dims (must match reference)
#define LNUM  6
#define DIM   512
#define HIDD  1024
#define VOC   32000
#define SEQ   512     // S == T
#define NH    4
#define DKH   128
#define BATCH 4

typedef __attribute__((ext_vector_type(16))) _Float16 v16h;
typedef __attribute__((ext_vector_type(8)))  _Float16 h8;
typedef __attribute__((ext_vector_type(8)))  float    v8f;

__device__ __forceinline__ float gelu_exact(float x) {
  return 0.5f * x * (1.0f + erff(x * 0.70710678118654752f));
}

// f16 fragment load (operand stored row-major along K, 16B aligned).
// p = row_base + k0 + half*8 ; halves [0..7] = K {0..7}, [8..15] = K {16..23}.
__device__ __forceinline__ v16h load_frag16(const _Float16* __restrict__ p) {
  const h8 lo = *(const h8*)(p);
  const h8 hi = *(const h8*)(p + 16);
  return __builtin_shufflevector(lo, hi, 0,1,2,3,4,5,6,7,8,9,10,11,12,13,14,15);
}

// ---------------------------------------------------------------------------
// Batched WMMA GEMM, all-f16 operands:  C = act( A @ B^T (+bias) (+Res) )
//   A : f16 [M,K] row-major (lda), batch offsets bb*sAb + hh*sAh
//   B : f16 [N,K] row-major (ldb)  (i.e. operand columns stored as rows)
//   Outputs (any subset): Cf (f32), Ch (f16, same geometry), Ct (f16 transposed
//   per-head copy: idx = (crow/tChunk)*tStride + ccol*ldt + crow%tChunk).
// One wave computes a 32x64 block = 2x4 grid of 16x16 tiles; per k-step:
// 12 b128 loads -> 8 v_wmma_f32_16x16x32_f16, zero conversions.
// M%32==0, N%64==0, K%32==0 for every GEMM here -> no edge handling; tile
// selection is wave-uniform -> EXEC all-ones for WMMA.
// ---------------------------------------------------------------------------
__global__ __launch_bounds__(128)
void gemm_wmma_kernel(const _Float16* __restrict__ A, long sAb, long sAh, int lda,
                      const _Float16* __restrict__ Bm, long sBb, long sBh, int ldb,
                      float* __restrict__ Cf, _Float16* __restrict__ Ch,
                      long sCb, long sCh, int ldc,
                      _Float16* __restrict__ Ct, int tChunk, long tStride, int ldt,
                      const float* __restrict__ bias,
                      const float* __restrict__ Res, int ldres,
                      int M, int N, int K, int nh, int act)
{
  const int bh = blockIdx.y;
  const int bb = bh / nh, hh = bh % nh;
  A  += (long)bb * sAb + (long)hh * sAh;
  Bm += (long)bb * sBb + (long)hh * sBh;
  const long coff = (long)bb * sCb + (long)hh * sCh;

  const int  tilesN     = N >> 6;                  // 64-wide macro tiles
  const long tilesTotal = (long)(M >> 5) * tilesN; // 32-tall macro tiles
  const long tile = (long)blockIdx.x * (blockDim.x >> 5) + (threadIdx.x >> 5);
  if (tile >= tilesTotal) return;   // wave-uniform exit

  const int tm   = (int)(tile / tilesN);
  const int tn   = (int)(tile % tilesN);
  const int lane = threadIdx.x & 31;
  const int half = lane >> 4;       // K-base selector per ISA 16-bit A/B layout
  const int idx  = lane & 15;

  const _Float16* Ar[2];
#pragma unroll
  for (int i = 0; i < 2; ++i)
    Ar[i] = A + (long)(tm * 32 + i * 16 + idx) * lda + (half << 3);

  const _Float16* Br[4];
#pragma unroll
  for (int j = 0; j < 4; ++j)
    Br[j] = Bm + (long)(tn * 64 + j * 16 + idx) * ldb + (half << 3);

  v8f acc[2][4];
#pragma unroll
  for (int i = 0; i < 2; ++i)
#pragma unroll
    for (int j = 0; j < 4; ++j) acc[i][j] = (v8f){};

  for (int k0 = 0; k0 < K; k0 += 32) {
    v16h af[2], bf[4];
#pragma unroll
    for (int i = 0; i < 2; ++i) af[i] = load_frag16(Ar[i] + k0);
#pragma unroll
    for (int j = 0; j < 4; ++j) bf[j] = load_frag16(Br[j] + k0);
#pragma unroll
    for (int i = 0; i < 2; ++i)
#pragma unroll
      for (int j = 0; j < 4; ++j)
        acc[i][j] = __builtin_amdgcn_wmma_f32_16x16x32_f16(
            false, af[i], false, bf[j], (short)0, acc[i][j], false, false);
  }

  // C/D layout: VGPR r -> lanes0-15: M=r, lanes16-31: M=r+8; N = idx
#pragma unroll
  for (int j = 0; j < 4; ++j) {
    const int   ccol = tn * 64 + j * 16 + idx;
    const float bv   = bias ? bias[ccol] : 0.0f;
#pragma unroll
    for (int i = 0; i < 2; ++i) {
#pragma unroll
      for (int r = 0; r < 8; ++r) {
        const int crow = tm * 32 + i * 16 + (half << 3) + r;
        float val = acc[i][j][r] + bv;
        if (Res) val += Res[(long)crow * ldres + ccol];
        if (act) val = gelu_exact(val);
        if (Cf) Cf[coff + (long)crow * ldc + ccol] = val;
        if (Ch) Ch[coff + (long)crow * ldc + ccol] = (_Float16)val;
        if (Ct) Ct[(long)(crow / tChunk) * tStride + (long)ccol * ldt + (crow % tChunk)]
                  = (_Float16)val;
      }
    }
  }
}

// ---------------------------------------------------------------------------
// Weight prep: W[K,N] f32 -> Wt[N,K] f16 (LDS-tiled transpose + convert).
// grid = (N/16, K/16, L), block = 256 (16x16).
// ---------------------------------------------------------------------------
__global__ __launch_bounds__(256)
void wconv_kernel(const float* __restrict__ W, _Float16* __restrict__ Wt, int K, int N)
{
  __shared__ float tile[16][17];
  const long lo = (long)blockIdx.z * K * N;
  const int  n0 = blockIdx.x * 16, k0 = blockIdx.y * 16;
  const int  tx = threadIdx.x & 15, ty = threadIdx.x >> 4;
  tile[ty][tx] = W[lo + (long)(k0 + ty) * N + (n0 + tx)];
  __syncthreads();
  Wt[lo + (long)(n0 + ty) * K + (k0 + tx)] = (_Float16)tile[tx][ty];
}

// ---------------------------------------------------------------------------
// Embedding gather + positional add (f32 stream + f16 GEMM copy)
// ---------------------------------------------------------------------------
__global__ __launch_bounds__(256)
void embed_kernel(const int* __restrict__ tok, const float* __restrict__ emb,
                  const float* __restrict__ pos, float* __restrict__ out,
                  _Float16* __restrict__ outh, int seqlen)
{
  const int  row = blockIdx.x;          // b*seqlen + s
  const int  s   = row % seqlen;
  const long t   = (long)tok[row];
  for (int d = threadIdx.x; d < DIM; d += blockDim.x) {
    const float val = emb[t * DIM + d] + pos[(long)s * DIM + d];
    out [(long)row * DIM + d] = val;
    outh[(long)row * DIM + d] = (_Float16)val;
  }
}

// ---------------------------------------------------------------------------
// LayerNorm over last dim (DIM=512): one wave per row; writes f32 + f16 copies.
// ---------------------------------------------------------------------------
__global__ __launch_bounds__(256)
void layernorm_kernel(const float* __restrict__ x, const float* __restrict__ sc,
                      const float* __restrict__ bi, float* __restrict__ y,
                      _Float16* __restrict__ yh, int rows)
{
  const int row = blockIdx.x * (blockDim.x >> 5) + (threadIdx.x >> 5);
  if (row >= rows) return;
  const int lane = threadIdx.x & 31;
  const float* xr = x + (long)row * DIM;

  float vals[DIM / 32];
  float s = 0.0f;
#pragma unroll
  for (int i = 0; i < DIM / 32; ++i) { vals[i] = xr[lane + i * 32]; s += vals[i]; }
#pragma unroll
  for (int off = 16; off > 0; off >>= 1) s += __shfl_xor(s, off, 32);
  const float mean = s * (1.0f / DIM);

  float v = 0.0f;
#pragma unroll
  for (int i = 0; i < DIM / 32; ++i) { const float d = vals[i] - mean; v += d * d; }
#pragma unroll
  for (int off = 16; off > 0; off >>= 1) v += __shfl_xor(v, off, 32);
  const float inv = rsqrtf(v * (1.0f / DIM) + 1e-5f);

#pragma unroll
  for (int i = 0; i < DIM / 32; ++i) {
    const int c = lane + i * 32;
    const float o = (vals[i] - mean) * inv * sc[c] + bi[c];
    y [(long)row * DIM + c] = o;
    yh[(long)row * DIM + c] = (_Float16)o;
  }
}

// ---------------------------------------------------------------------------
// Masked softmax over keys (len SEQ): reads f32 scores, writes f16 probs.
// ---------------------------------------------------------------------------
__global__ __launch_bounds__(256)
void softmax_kernel(const float* __restrict__ scores, _Float16* __restrict__ probs,
                    const int* __restrict__ pad, int causal, float scale)
{
  const long row  = (long)blockIdx.x * (blockDim.x >> 5) + (threadIdx.x >> 5);
  const int  lane = threadIdx.x & 31;
  const int  b    = (int)(row / ((long)NH * SEQ));
  const int  qi   = (int)(row % SEQ);
  const float* r = scores + row * SEQ;
  _Float16*    p = probs  + row * SEQ;
  const int*  pb = pad + (long)b * SEQ;

  float vals[SEQ / 32];
  float m = -INFINITY;
#pragma unroll
  for (int i = 0; i < SEQ / 32; ++i) {
    const int kc = lane + i * 32;
    float vv = r[kc] * scale;
    const bool ok = (pb[kc] != 0) && (!causal || (kc <= qi));
    vv = ok ? vv : -INFINITY;
    vals[i] = vv;
    m = fmaxf(m, vv);
  }
#pragma unroll
  for (int off = 16; off > 0; off >>= 1) m = fmaxf(m, __shfl_xor(m, off, 32));

  float sum = 0.0f;
#pragma unroll
  for (int i = 0; i < SEQ / 32; ++i) { const float e = __expf(vals[i] - m); vals[i] = e; sum += e; }
#pragma unroll
  for (int off = 16; off > 0; off >>= 1) sum += __shfl_xor(sum, off, 32);
  const float inv = 1.0f / sum;
#pragma unroll
  for (int i = 0; i < SEQ / 32; ++i) p[lane + i * 32] = (_Float16)(vals[i] * inv);
}

// ---------------------------------------------------------------------------
// Host-side GEMM launcher (32x64 macro tiles, 4 waves per block)
// ---------------------------------------------------------------------------
static void ggemm(hipStream_t st,
                  const _Float16* A, long sAb, long sAh, int lda,
                  const _Float16* Bm, long sBb, long sBh, int ldb,
                  float* Cf, _Float16* Ch, long sCb, long sCh, int ldc,
                  _Float16* Ct, int tChunk, long tStride, int ldt,
                  const float* bias, const float* Res, int ldres,
                  int M, int N, int K, int nb, int nh, int act)
{
  const long tiles = (long)(M / 32) * (N / 64);
  dim3 grid((unsigned)((tiles + 3) / 4), (unsigned)(nb * nh));
  dim3 block(128);
  gemm_wmma_kernel<<<grid, block, 0, st>>>(A, sAb, sAh, lda, Bm, sBb, sBh, ldb,
      Cf, Ch, sCb, sCh, ldc, Ct, tChunk, tStride, ldt, bias, Res, ldres,
      M, N, K, nh, act);
}

static void wconv(hipStream_t st, const float* W, _Float16* Wt, int K, int N, int nl) {
  dim3 grid(N / 16, K / 16, nl);
  wconv_kernel<<<grid, 256, 0, st>>>(W, Wt, K, N);
}

extern "C" void kernel_launch(void* const* d_in, const int* in_sizes, int n_in,
                              void* d_out, int out_size, void* d_ws, size_t ws_size,
                              hipStream_t stream) {
  // ---- inputs (setup_inputs() dict order, params flattened in insertion order)
  const int*   src  = (const int*)d_in[0];
  const int*   tgt  = (const int*)d_in[1];
  const int*   spad = (const int*)d_in[2];
  const int*   tpad = (const int*)d_in[3];
  const float* emb_src = (const float*)d_in[4];
  const float* emb_tgt = (const float*)d_in[5];
  const float* pos_src = (const float*)d_in[6];
  const float* pos_tgt = (const float*)d_in[7];
  const float* eWq = (const float*)d_in[8];   const float* ebq = (const float*)d_in[9];
  const float* eWk = (const float*)d_in[10];  const float* ebk = (const float*)d_in[11];
  const float* eWv = (const float*)d_in[12];  const float* ebv = (const float*)d_in[13];
  const float* eWo = (const float*)d_in[14];  const float* ebo = (const float*)d_in[15];
  const float* eln1s = (const float*)d_in[16]; const float* eln1b = (const float*)d_in[17];
  const float* eln2s = (const float*)d_in[18]; const float* eln2b = (const float*)d_in[19];
  const float* eW1 = (const float*)d_in[20];  const float* eb1 = (const float*)d_in[21];
  const float* eW2 = (const float*)d_in[22];  const float* eb2 = (const float*)d_in[23];
  const float* sWq = (const float*)d_in[24];  const float* sbq = (const float*)d_in[25];
  const float* sWk = (const float*)d_in[26];  const float* sbk = (const float*)d_in[27];
  const float* sWv = (const float*)d_in[28];  const float* sbv = (const float*)d_in[29];
  const float* sWo = (const float*)d_in[30];  const float* sbo = (const float*)d_in[31];
  const float* cWq = (const float*)d_in[32];  const float* cbq = (const float*)d_in[33];
  const float* cWk = (const float*)d_in[34];  const float* cbk = (const float*)d_in[35];
  const float* cWv = (const float*)d_in[36];  const float* cbv = (const float*)d_in[37];
  const float* cWo = (const float*)d_in[38];  const float* cbo = (const float*)d_in[39];
  const float* dln1s = (const float*)d_in[40]; const float* dln1b = (const float*)d_in[41];
  const float* dln2s = (const float*)d_in[42]; const float* dln2b = (const float*)d_in[43];
  const float* dln3s = (const float*)d_in[44]; const float* dln3b = (const float*)d_in[45];
  const float* dW1 = (const float*)d_in[46];  const float* db1 = (const float*)d_in[47];
  const float* dW2 = (const float*)d_in[48];  const float* db2 = (const float*)d_in[49];
  const float* ffnW = (const float*)d_in[50]; const float* ffnb = (const float*)d_in[51];

  // ---- workspace layout
  const long BSD = (long)BATCH * SEQ * DIM;
  const long BHSS = (long)BATCH * NH * SEQ * SEQ;
  float* fw = (float*)d_ws;
  float* xe = fw; fw += BSD;   // encoder stream (f32) / final memory
  float* xd = fw; fw += BSD;   // decoder stream (f32)
  float* t1 = fw; fw += BSD;
  float* t2 = fw; fw += BSD;
  float* h1 = fw; fw += BSD;
  float* h2 = fw; fw += BSD;
  float* sc = fw; fw += BHSS;  // attention scores (f32)

  _Float16* hw = (_Float16*)fw;
  _Float16* xeh = hw; hw += BSD;   // f16 copies feeding GEMM A-operands
  _Float16* xdh = hw; hw += BSD;
  _Float16* qh  = hw; hw += BSD;
  _Float16* kh  = hw; hw += BSD;
  _Float16* aoh = hw; hw += BSD;
  _Float16* h1h = hw; hw += BSD;
  _Float16* h2h = hw; hw += BSD;
  _Float16* vt  = hw; hw += BSD;                       // V^T per head: [b][h][dk][s]
  _Float16* midh = hw; hw += (long)BATCH * SEQ * HIDD;
  _Float16* ph  = hw; hw += BHSS;                      // softmax probs (f16)
  // f16 transposed weight arena [N,K]
  const long DD = (long)DIM * DIM, DH = (long)DIM * HIDD;
  _Float16* wEq = hw; hw += LNUM * DD;  _Float16* wEk = hw; hw += LNUM * DD;
  _Float16* wEv = hw; hw += LNUM * DD;  _Float16* wEo = hw; hw += LNUM * DD;
  _Float16* wE1 = hw; hw += LNUM * DH;  _Float16* wE2 = hw; hw += LNUM * DH;
  _Float16* wSq = hw; hw += LNUM * DD;  _Float16* wSk = hw; hw += LNUM * DD;
  _Float16* wSv = hw; hw += LNUM * DD;  _Float16* wSo = hw; hw += LNUM * DD;
  _Float16* wCq = hw; hw += LNUM * DD;  _Float16* wCk = hw; hw += LNUM * DD;
  _Float16* wCv = hw; hw += LNUM * DD;  _Float16* wCo = hw; hw += LNUM * DD;
  _Float16* wD1 = hw; hw += LNUM * DH;  _Float16* wD2 = hw; hw += LNUM * DH;
  _Float16* wF  = hw; hw += (long)DIM * VOC;

  const int   M2    = BATCH * SEQ;                 // 2048 rows
  const float isqdk = 0.08838834764831845f;        // 1/sqrt(128)
  const long  sSC_b = (long)NH * SEQ * SEQ, sSC_h = (long)SEQ * SEQ;
  const long  sXD_b = (long)SEQ * DIM;
  const long  sVT_b = (long)DIM * SEQ, sVT_h = (long)DKH * SEQ;
  const int   smBlocks = (BATCH * NH * SEQ) / 8;
  const int   lnBlocks = M2 / 8;

  // ---- weight prep: f32 [K,N] -> f16 [N,K], all layers at once (grid.z = L)
  wconv(stream, eWq, wEq, DIM, DIM, LNUM);  wconv(stream, eWk, wEk, DIM, DIM, LNUM);
  wconv(stream, eWv, wEv, DIM, DIM, LNUM);  wconv(stream, eWo, wEo, DIM, DIM, LNUM);
  wconv(stream, eW1, wE1, DIM, HIDD, LNUM); wconv(stream, eW2, wE2, HIDD, DIM, LNUM);
  wconv(stream, sWq, wSq, DIM, DIM, LNUM);  wconv(stream, sWk, wSk, DIM, DIM, LNUM);
  wconv(stream, sWv, wSv, DIM, DIM, LNUM);  wconv(stream, sWo, wSo, DIM, DIM, LNUM);
  wconv(stream, cWq, wCq, DIM, DIM, LNUM);  wconv(stream, cWk, wCk, DIM, DIM, LNUM);
  wconv(stream, cWv, wCv, DIM, DIM, LNUM);  wconv(stream, cWo, wCo, DIM, DIM, LNUM);
  wconv(stream, dW1, wD1, DIM, HIDD, LNUM); wconv(stream, dW2, wD2, HIDD, DIM, LNUM);
  wconv(stream, ffnW, wF, DIM, VOC, 1);

  // ---- embeddings + positions
  embed_kernel<<<BATCH * SEQ, 256, 0, stream>>>(src, emb_src, pos_src, xe, xeh, SEQ);
  embed_kernel<<<BATCH * SEQ, 256, 0, stream>>>(tgt, emb_tgt, pos_tgt, xd, xdh, SEQ);

  // ---- encoder
  for (int l = 0; l < LNUM; ++l) {
    const long oDD = (long)l * DD, oD = (long)l * DIM, oDH = (long)l * DH, oH = (long)l * HIDD;
    // projections -> f16 (q,k) and transposed f16 (v)
    ggemm(stream, xeh,0,0,DIM, wEq+oDD,0,0,DIM, nullptr,qh,0,0,DIM, nullptr,1,0,1,
          ebq+oD, nullptr,0, M2,DIM,DIM, 1,1, 0);
    ggemm(stream, xeh,0,0,DIM, wEk+oDD,0,0,DIM, nullptr,kh,0,0,DIM, nullptr,1,0,1,
          ebk+oD, nullptr,0, M2,DIM,DIM, 1,1, 0);
    ggemm(stream, xeh,0,0,DIM, wEv+oDD,0,0,DIM, nullptr,nullptr,0,0,DIM, vt,SEQ,sVT_b,SEQ,
          ebv+oD, nullptr,0, M2,DIM,DIM, 1,1, 0);
    // scores = Q @ K^T  (per b,h) -> f32
    ggemm(stream, qh,sXD_b,DKH,DIM, kh,sXD_b,DKH,DIM, sc,nullptr,sSC_b,sSC_h,SEQ,
          nullptr,1,0,1, nullptr, nullptr,0, SEQ,SEQ,DKH, BATCH,NH, 0);
    softmax_kernel<<<smBlocks, 256, 0, stream>>>(sc, ph, spad, 0, isqdk);
    // out = P @ V   (B operand = V^T rows) -> f16 merged heads
    ggemm(stream, ph,sSC_b,sSC_h,SEQ, vt,sVT_b,sVT_h,SEQ, nullptr,aoh,sXD_b,DKH,DIM,
          nullptr,1,0,1, nullptr, nullptr,0, SEQ,DKH,SEQ, BATCH,NH, 0);
    // o-proj + residual -> f32, LN1 -> h1 (f32) + h1h (f16)
    ggemm(stream, aoh,0,0,DIM, wEo+oDD,0,0,DIM, t1,nullptr,0,0,DIM, nullptr,1,0,1,
          ebo+oD, xe,DIM, M2,DIM,DIM, 1,1, 0);
    layernorm_kernel<<<lnBlocks, 256, 0, stream>>>(t1, eln1s+oD, eln1b+oD, h1, h1h, M2);
    // FFN: gelu(h1@W1+b1) -> f16 ; @W2+b2 + h1 -> f32 ; LN2 -> xe + xeh
    ggemm(stream, h1h,0,0,DIM, wE1+oDH,0,0,DIM, nullptr,midh,0,0,HIDD, nullptr,1,0,1,
          eb1+oH, nullptr,0, M2,HIDD,DIM, 1,1, 1);
    ggemm(stream, midh,0,0,HIDD, wE2+oDH,0,0,HIDD, t2,nullptr,0,0,DIM, nullptr,1,0,1,
          eb2+oD, h1,DIM, M2,DIM,HIDD, 1,1, 0);
    layernorm_kernel<<<lnBlocks, 256, 0, stream>>>(t2, eln2s+oD, eln2b+oD, xe, xeh, M2);
  }

  // ---- decoder (memory = xe/xeh)
  for (int l = 0; l < LNUM; ++l) {
    const long oDD = (long)l * DD, oD = (long)l * DIM, oDH = (long)l * DH, oH = (long)l * HIDD;
    // self-attention (causal)
    ggemm(stream, xdh,0,0,DIM, wSq+oDD,0,0,DIM, nullptr,qh,0,0,DIM, nullptr,1,0,1,
          sbq+oD, nullptr,0, M2,DIM,DIM, 1,1, 0);
    ggemm(stream, xdh,0,0,DIM, wSk+oDD,0,0,DIM, nullptr,kh,0,0,DIM, nullptr,1,0,1,
          sbk+oD, nullptr,0, M2,DIM,DIM, 1,1, 0);
    ggemm(stream, xdh,0,0,DIM, wSv+oDD,0,0,DIM, nullptr,nullptr,0,0,DIM, vt,SEQ,sVT_b,SEQ,
          sbv+oD, nullptr,0, M2,DIM,DIM, 1,1, 0);
    ggemm(stream, qh,sXD_b,DKH,DIM, kh,sXD_b,DKH,DIM, sc,nullptr,sSC_b,sSC_h,SEQ,
          nullptr,1,0,1, nullptr, nullptr,0, SEQ,SEQ,DKH, BATCH,NH, 0);
    softmax_kernel<<<smBlocks, 256, 0, stream>>>(sc, ph, tpad, 1, isqdk);
    ggemm(stream, ph,sSC_b,sSC_h,SEQ, vt,sVT_b,sVT_h,SEQ, nullptr,aoh,sXD_b,DKH,DIM,
          nullptr,1,0,1, nullptr, nullptr,0, SEQ,DKH,SEQ, BATCH,NH, 0);
    ggemm(stream, aoh,0,0,DIM, wSo+oDD,0,0,DIM, t1,nullptr,0,0,DIM, nullptr,1,0,1,
          sbo+oD, xd,DIM, M2,DIM,DIM, 1,1, 0);
    layernorm_kernel<<<lnBlocks, 256, 0, stream>>>(t1, dln1s+oD, dln1b+oD, h1, h1h, M2);
    // cross-attention (Q from h1h, K/V from memory xeh)
    ggemm(stream, h1h,0,0,DIM, wCq+oDD,0,0,DIM, nullptr,qh,0,0,DIM, nullptr,1,0,1,
          cbq+oD, nullptr,0, M2,DIM,DIM, 1,1, 0);
    ggemm(stream, xeh,0,0,DIM, wCk+oDD,0,0,DIM, nullptr,kh,0,0,DIM, nullptr,1,0,1,
          cbk+oD, nullptr,0, M2,DIM,DIM, 1,1, 0);
    ggemm(stream, xeh,0,0,DIM, wCv+oDD,0,0,DIM, nullptr,nullptr,0,0,DIM, vt,SEQ,sVT_b,SEQ,
          cbv+oD, nullptr,0, M2,DIM,DIM, 1,1, 0);
    ggemm(stream, qh,sXD_b,DKH,DIM, kh,sXD_b,DKH,DIM, sc,nullptr,sSC_b,sSC_h,SEQ,
          nullptr,1,0,1, nullptr, nullptr,0, SEQ,SEQ,DKH, BATCH,NH, 0);
    softmax_kernel<<<smBlocks, 256, 0, stream>>>(sc, ph, spad, 0, isqdk);
    ggemm(stream, ph,sSC_b,sSC_h,SEQ, vt,sVT_b,sVT_h,SEQ, nullptr,aoh,sXD_b,DKH,DIM,
          nullptr,1,0,1, nullptr, nullptr,0, SEQ,DKH,SEQ, BATCH,NH, 0);
    ggemm(stream, aoh,0,0,DIM, wCo+oDD,0,0,DIM, t2,nullptr,0,0,DIM, nullptr,1,0,1,
          cbo+oD, h1,DIM, M2,DIM,DIM, 1,1, 0);
    layernorm_kernel<<<lnBlocks, 256, 0, stream>>>(t2, dln2s+oD, dln2b+oD, h2, h2h, M2);
    // FFN + residual, LN3 -> xd + xdh
    ggemm(stream, h2h,0,0,DIM, wD1+oDH,0,0,DIM, nullptr,midh,0,0,HIDD, nullptr,1,0,1,
          db1+oH, nullptr,0, M2,HIDD,DIM, 1,1, 1);
    ggemm(stream, midh,0,0,HIDD, wD2+oDH,0,0,HIDD, t1,nullptr,0,0,DIM, nullptr,1,0,1,
          db2+oD, h2,DIM, M2,DIM,HIDD, 1,1, 0);
    layernorm_kernel<<<lnBlocks, 256, 0, stream>>>(t1, dln3s+oD, dln3b+oD, xd, xdh, M2);
  }

  // ---- final logits: [2048,512] f16 @ wF[V,D] f16 + bias -> d_out (f32)
  ggemm(stream, xdh,0,0,DIM, wF,0,0,DIM, (float*)d_out,nullptr,0,0,VOC, nullptr,1,0,1,
        ffnb, nullptr,0, M2,VOC,DIM, 1,1, 0);
}
